// TransformerCrossAttnLayer_37555194036289
// MI455X (gfx1250) — matrix operand order repaired and checked by reading, not compiled, not run on hardware
//
#include <hip/hip_runtime.h>

typedef __bf16 bf16;
typedef __attribute__((ext_vector_type(16))) bf16 v16bf;
typedef __attribute__((ext_vector_type(8)))  bf16 v8bf;
typedef __attribute__((ext_vector_type(8)))  float v8f;

#define WMMA_BF16(a, b, c) \
  __builtin_amdgcn_wmma_f32_16x16x32_bf16(false, (a), false, (b), (short)0, (c), false, false)

// ---------------------------------------------------------------------------
// Fragment loaders (wave32 WMMA VGPR striping, CDNA5 ISA 7.12.2)
// A (16x32 bf16): lane m=l%16, h=l/16. elements 0..7 -> K=8h+0..7, 8..15 -> K=16+8h+0..7
// B (32x16 bf16): lane n=l%16, h=l/16. elements 0..7 -> K=8h+0..7, 8..15 -> K=16+8h+0..7
// C/D (16x16 f32): lane col=l%16; VGPR r -> row r+8h
// ---------------------------------------------------------------------------
__device__ __forceinline__ v16bf frag_a(const bf16* __restrict__ p, int h) {
  const v8bf* q = (const v8bf*)(p + 8 * h);
  v8bf lo = q[0], hi = q[2];
  v16bf r;
#pragma unroll
  for (int i = 0; i < 8; ++i) { r[i] = lo[i]; r[i + 8] = hi[i]; }
  return r;
}
__device__ __forceinline__ v16bf frag_b(const bf16* __restrict__ p, int h) {
  const v8bf* q = (const v8bf*)(p + 8 * h);
  v8bf lo = q[0], hi = q[2];
  v16bf r;
#pragma unroll
  for (int i = 0; i < 8; ++i) { r[i] = lo[i]; r[i + 8] = hi[i]; }
  return r;
}
__device__ __forceinline__ v8f vzero8() {
  v8f z = {0.f, 0.f, 0.f, 0.f, 0.f, 0.f, 0.f, 0.f};
  return z;
}
__device__ __forceinline__ float wave_sum(float v) {
#pragma unroll
  for (int off = 16; off > 0; off >>= 1) v += __shfl_xor(v, off, 32);
  return v;
}
__device__ __forceinline__ float wave_max(float v) {
#pragma unroll
  for (int off = 16; off > 0; off >>= 1) v = fmaxf(v, __shfl_xor(v, off, 32));
  return v;
}

// ---------------------------------------------------------------------------
// f32 -> bf16 convert
// ---------------------------------------------------------------------------
__global__ void k_cvt_bf16(const float* __restrict__ src, bf16* __restrict__ dst, int n) {
  int i = blockIdx.x * blockDim.x + threadIdx.x;
  if (i < n) dst[i] = (bf16)src[i];
}

// ---------------------------------------------------------------------------
// LayerNorm over C=256; one wave per row; 8 rows per 256-thread block.
// remap!=0: source row r lives at output-concat row (r/64)*128 + 64 + r%64.
// ---------------------------------------------------------------------------
__global__ void k_layernorm(const float* __restrict__ x, const float* __restrict__ g,
                            const float* __restrict__ b, bf16* __restrict__ out, int remap) {
  int wave = threadIdx.x >> 5, lane = threadIdx.x & 31;
  int row = blockIdx.x * 8 + wave;
  int srow = remap ? (((row >> 6) << 7) + 64 + (row & 63)) : row;
  const float* xr = x + (size_t)srow * 256 + lane * 8;
  float v[8];
#pragma unroll
  for (int j = 0; j < 8; ++j) v[j] = xr[j];
  float s = 0.f;
#pragma unroll
  for (int j = 0; j < 8; ++j) s += v[j];
  s = wave_sum(s);
  float mean = s * (1.0f / 256.0f);
  float s2 = 0.f;
#pragma unroll
  for (int j = 0; j < 8; ++j) { float d = v[j] - mean; s2 += d * d; }
  s2 = wave_sum(s2);
  float rstd = rsqrtf(s2 * (1.0f / 256.0f) + 1e-5f);
  bf16* orow = out + (size_t)row * 256 + lane * 8;
#pragma unroll
  for (int j = 0; j < 8; ++j) {
    int c = lane * 8 + j;
    orow[j] = (bf16)((v[j] - mean) * rstd * g[c] + b[c]);
  }
}

// ---------------------------------------------------------------------------
// out[m,col] = ((sum_k A[m,k] * Wt[col,k]) + bias[col]) * scale  (bf16 out)
// A: [M,256] bf16 row-major, Wt: [Ncols,256] bf16 row-major (i.e. B^T).
// grid (M/16, Ncols/64), block 128 (4 waves, one 16x16 tile each per column group)
// ---------------------------------------------------------------------------
__global__ void k_proj_gemm(const bf16* __restrict__ A, const bf16* __restrict__ Wt,
                            const float* __restrict__ bias, float scale,
                            bf16* __restrict__ out, int ldo) {
  int wave = threadIdx.x >> 5, lane = threadIdx.x & 31;
  int q = lane & 15, h = lane >> 4;
  int m0 = blockIdx.x * 16;
  int n0 = blockIdx.y * 64 + wave * 16;
  const bf16* arow = A + (size_t)(m0 + q) * 256;
  const bf16* brow = Wt + (size_t)(n0 + q) * 256;
  v8f acc = vzero8();
#pragma unroll
  for (int k0 = 0; k0 < 256; k0 += 32) {
    v16bf a = frag_a(arow + k0, h);
    v16bf b = frag_b(brow + k0, h);
    acc = WMMA_BF16(a, b, acc);
  }
  int col = n0 + q;
  float bv = bias[col];
#pragma unroll
  for (int r = 0; r < 8; ++r) {
    int m = m0 + r + 8 * h;
    out[(size_t)m * ldo + col] = (bf16)((acc[r] + bv) * scale);
  }
}

// ---------------------------------------------------------------------------
// Output projection + bias + residual, writes f32 into the strided [W,2N,C]
// output (half selected by halfoff = 0 or 64). grid (1024,4), block 128.
// ---------------------------------------------------------------------------
__global__ void k_outproj_gemm(const bf16* __restrict__ A, const bf16* __restrict__ Wt,
                               const float* __restrict__ bias, const float* __restrict__ res,
                               float* __restrict__ dout, int halfoff) {
  int wave = threadIdx.x >> 5, lane = threadIdx.x & 31;
  int q = lane & 15, h = lane >> 4;
  int m0 = blockIdx.x * 16;
  int n0 = blockIdx.y * 64 + wave * 16;
  const bf16* arow = A + (size_t)(m0 + q) * 256;
  const bf16* brow = Wt + (size_t)(n0 + q) * 256;
  v8f acc = vzero8();
#pragma unroll
  for (int k0 = 0; k0 < 256; k0 += 32) {
    v16bf a = frag_a(arow + k0, h);
    v16bf b = frag_b(brow + k0, h);
    acc = WMMA_BF16(a, b, acc);
  }
  int col = n0 + q;
  float bv = bias[col];
#pragma unroll
  for (int r = 0; r < 8; ++r) {
    int m = m0 + r + 8 * h;
    int g = ((m >> 6) << 7) + halfoff + (m & 63);
    dout[(size_t)g * 256 + col] = acc[r] + bv + res[(size_t)m * 256 + col];
  }
}

// ---------------------------------------------------------------------------
// Score term 2: raw[n,e,w,v] = q[w,n,e,:] . Pk[idx(w,v), e-slice]   (writes '=')
// grid (W=256, H=8), block 128; wave = n-tile of 16; loop v-tiles.
// ---------------------------------------------------------------------------
__global__ void k_term2(const bf16* __restrict__ qry, const bf16* __restrict__ Pk,
                        const int* __restrict__ pidx, int flip, float* __restrict__ raw) {
  int w = blockIdx.x, e = blockIdx.y;
  int wave = threadIdx.x >> 5, lane = threadIdx.x & 31;
  int q = lane & 15, h = lane >> 4;
  int n0 = wave * 16;
  v16bf a = frag_a(qry + (size_t)(w * 64 + n0 + q) * 256 + e * 32, h);
  for (int v0 = 0; v0 < 256; v0 += 16) {
    int s = pidx[w * 256 + v0 + q];
    s = flip ? (510 - s) : s;
    v16bf b = frag_b(Pk + (size_t)s * 256 + e * 32, h);
    v8f acc = vzero8();
    acc = WMMA_BF16(a, b, acc);
#pragma unroll
    for (int r = 0; r < 8; ++r)
      raw[(size_t)((n0 + 8 * h + r) * 8 + e) * 65536 + w * 256 + v0 + q] = acc[r];
  }
}

// ---------------------------------------------------------------------------
// Score term 3: raw[n,e,w,v] += k[v,n,e,:] . Pq[idx(w,v), e-slice]
// grid (W'=256, H=8), block 128; wave = n-tile; loop w-tiles; C-operand = raw.
// ---------------------------------------------------------------------------
__global__ void k_term3(const bf16* __restrict__ kmat, const bf16* __restrict__ Pq,
                        const int* __restrict__ pidx, int flip, float* __restrict__ raw) {
  int v = blockIdx.x, e = blockIdx.y;
  int wave = threadIdx.x >> 5, lane = threadIdx.x & 31;
  int q = lane & 15, h = lane >> 4;
  int n0 = wave * 16;
  v16bf a = frag_a(kmat + (size_t)(v * 64 + n0 + q) * 256 + e * 32, h);
  for (int w0 = 0; w0 < 256; w0 += 16) {
    int s = pidx[(w0 + q) * 256 + v];
    s = flip ? (510 - s) : s;
    v16bf b = frag_b(Pq + (size_t)s * 256 + e * 32, h);
    v8f c;
#pragma unroll
    for (int r = 0; r < 8; ++r)
      c[r] = raw[(size_t)((n0 + 8 * h + r) * 8 + e) * 65536 + (w0 + q) * 256 + v];
    v8f acc = WMMA_BF16(a, b, c);
#pragma unroll
    for (int r = 0; r < 8; ++r)
      raw[(size_t)((n0 + 8 * h + r) * 8 + e) * 65536 + (w0 + q) * 256 + v] = acc[r];
  }
}

// ---------------------------------------------------------------------------
// Score term 1: raw[n,e,w,v] += q[w,n,e,:] . k[v,n,e,:]
// grid (N=64, H=8), block 128; wave handles 4 w-tiles; C-operand = raw.
// ---------------------------------------------------------------------------
__global__ void k_term1(const bf16* __restrict__ qry, const bf16* __restrict__ kmat,
                        float* __restrict__ raw) {
  int n = blockIdx.x, e = blockIdx.y;
  int wave = threadIdx.x >> 5, lane = threadIdx.x & 31;
  int q = lane & 15, h = lane >> 4;
  for (int j = 0; j < 4; ++j) {
    int w0 = (wave * 4 + j) * 16;
    v16bf a = frag_a(qry + (size_t)((w0 + q) * 64 + n) * 256 + e * 32, h);
    for (int v0 = 0; v0 < 256; v0 += 16) {
      v16bf b = frag_b(kmat + (size_t)((v0 + q) * 64 + n) * 256 + e * 32, h);
      v8f c;
#pragma unroll
      for (int r = 0; r < 8; ++r)
        c[r] = raw[(size_t)(n * 8 + e) * 65536 + (w0 + 8 * h + r) * 256 + v0 + q];
      v8f acc = WMMA_BF16(a, b, c);
#pragma unroll
      for (int r = 0; r < 8; ++r)
        raw[(size_t)(n * 8 + e) * 65536 + (w0 + 8 * h + r) * 256 + v0 + q] = acc[r];
    }
  }
}

// ---------------------------------------------------------------------------
// out2[n,w,v] = sum_e raw[n,e,w,v]  (raw attention, summed over heads)
// grid 64*256 blocks, 256 threads.
// ---------------------------------------------------------------------------
__global__ void k_rawsum(const float* __restrict__ raw, float* __restrict__ out2) {
  int nw = blockIdx.x;
  int n = nw >> 8, w = nw & 255;
  int v = threadIdx.x;
  float s = 0.f;
#pragma unroll
  for (int e = 0; e < 8; ++e) s += raw[(size_t)(n * 8 + e) * 65536 + w * 256 + v];
  out2[(size_t)nw * 256 + v] = s;
}

// ---------------------------------------------------------------------------
// Fused softmax(raw) @ V -> vo[w,n,e*32+d] bf16.
// grid (64, 8, 2), block 64 (2 waves). V^T slab + per-wave attn tile in LDS.
// ---------------------------------------------------------------------------
__global__ void k_softmax_av(const float* __restrict__ raw, const bf16* __restrict__ vmat,
                             bf16* __restrict__ vo) {
  int n = blockIdx.x, e = blockIdx.y, zz = blockIdx.z;
  int wave = threadIdx.x >> 5, lane = threadIdx.x & 31;
  int q = lane & 15, h = lane >> 4;
  __shared__ bf16 VT[32][264];     // [d][v]
  __shared__ bf16 AT[2][16][264];  // per-wave attn tile [w-row][v]

  for (int rv = threadIdx.x; rv < 256; rv += 64) {
    const bf16* src = vmat + (size_t)(rv * 64 + n) * 256 + e * 32;
#pragma unroll
    for (int d = 0; d < 32; ++d) VT[d][rv] = src[d];
  }
  __syncthreads();

  for (int j = 0; j < 4; ++j) {
    int w0 = (zz * 8 + wave * 4 + j) * 16;
    // row-wise softmax, one full wave per row
    for (int rr = 0; rr < 16; ++rr) {
      const float* rrow = raw + (size_t)(n * 8 + e) * 65536 + (w0 + rr) * 256 + lane * 8;
      float x[8];
#pragma unroll
      for (int jj = 0; jj < 8; ++jj) x[jj] = rrow[jj];
      float mx = x[0];
#pragma unroll
      for (int jj = 1; jj < 8; ++jj) mx = fmaxf(mx, x[jj]);
      mx = wave_max(mx);
      float sm = 0.f;
#pragma unroll
      for (int jj = 0; jj < 8; ++jj) { x[jj] = __expf(x[jj] - mx); sm += x[jj]; }
      sm = wave_sum(sm);
      float inv = 1.0f / sm;
#pragma unroll
      for (int jj = 0; jj < 8; ++jj) AT[wave][rr][lane * 8 + jj] = (bf16)(x[jj] * inv);
    }
    __syncthreads();
    // attn(16x256) @ V(256x32) via 8 K-chunks of 32
    v8f acc0 = vzero8(), acc1 = vzero8();
#pragma unroll
    for (int v0 = 0; v0 < 256; v0 += 32) {
      v16bf a = frag_a(&AT[wave][q][v0], h);
      v16bf b0 = frag_b(&VT[q][v0], h);
      v16bf b1 = frag_b(&VT[16 + q][v0], h);
      acc0 = WMMA_BF16(a, b0, acc0);
      acc1 = WMMA_BF16(a, b1, acc1);
    }
#pragma unroll
    for (int r = 0; r < 8; ++r) {
      int w = w0 + r + 8 * h;
      bf16* dst = vo + (size_t)(w * 64 + n) * 256 + e * 32;
      dst[q] = (bf16)acc0[r];
      dst[16 + q] = (bf16)acc1[r];
    }
    __syncthreads();
  }
}

// ---------------------------------------------------------------------------
// Host side
// ---------------------------------------------------------------------------
namespace {
constexpr float kScale = 0.17677669529663687f;  // 32^-0.5

struct Ws {
  bf16 *lnL, *lnR, *lnT, *qry, *kmat, *vmat, *vo;
  bf16 *Pq, *Pk, *posb, *wbin, *wbout;
  float* raw;
};

void run_pass(const Ws& W, const bf16* qln, const bf16* kvln, int flip, int halfoff,
              const float* res, int do_rawsum, const float* in_b, const float* out_b,
              const int* pidx, float* out, hipStream_t stream) {
  const bf16* Wq = W.wbin;
  const bf16* Wk = W.wbin + 256 * 256;
  const bf16* Wv = W.wbin + 512 * 256;
  k_proj_gemm<<<dim3(1024, 4), 128, 0, stream>>>(qln, Wq, in_b, kScale, W.qry, 256);
  k_proj_gemm<<<dim3(1024, 4), 128, 0, stream>>>(kvln, Wk, in_b + 256, 1.0f, W.kmat, 256);
  k_proj_gemm<<<dim3(1024, 4), 128, 0, stream>>>(kvln, Wv, in_b + 512, 1.0f, W.vmat, 256);
  k_term2<<<dim3(256, 8), 128, 0, stream>>>(W.qry, W.Pk, pidx, flip, W.raw);
  k_term3<<<dim3(256, 8), 128, 0, stream>>>(W.kmat, W.Pq, pidx, flip, W.raw);
  k_term1<<<dim3(64, 8), 128, 0, stream>>>(W.qry, W.kmat, W.raw);
  if (do_rawsum)
    k_rawsum<<<64 * 256, 256, 0, stream>>>(W.raw, out + (size_t)256 * 128 * 256);
  k_softmax_av<<<dim3(64, 8, 2), 64, 0, stream>>>(W.raw, W.vmat, W.vo);
  k_outproj_gemm<<<dim3(1024, 4), 128, 0, stream>>>(W.vo, W.wbout, out_b, res, out, halfoff);
}
}  // namespace

extern "C" void kernel_launch(void* const* d_in, const int* in_sizes, int n_in,
                              void* d_out, int out_size, void* d_ws, size_t ws_size,
                              hipStream_t stream) {
  const float* feat_left = (const float*)d_in[0];
  const float* feat_right = (const float*)d_in[1];
  const float* pos = (const float*)d_in[2];
  const int* pidx = (const int*)d_in[3];
  const float* in_w = (const float*)d_in[4];
  const float* in_b = (const float*)d_in[5];
  const float* out_w = (const float*)d_in[6];
  const float* out_b = (const float*)d_in[7];
  const float* ln1_g = (const float*)d_in[8];
  const float* ln1_b = (const float*)d_in[9];
  const float* ln2_g = (const float*)d_in[10];
  const float* ln2_b = (const float*)d_in[11];
  float* out = (float*)d_out;

  char* base = (char*)d_ws;
  size_t o = 0;
  auto take = [&](size_t bytes) {
    char* p = base + o;
    o += (bytes + 255) & ~(size_t)255;
    return p;
  };
  Ws W;
  W.lnL = (bf16*)take((size_t)16384 * 256 * 2);
  W.lnR = (bf16*)take((size_t)16384 * 256 * 2);
  W.lnT = (bf16*)take((size_t)16384 * 256 * 2);
  W.qry = (bf16*)take((size_t)16384 * 256 * 2);
  W.kmat = (bf16*)take((size_t)16384 * 256 * 2);
  W.vmat = (bf16*)take((size_t)16384 * 256 * 2);
  W.vo = (bf16*)take((size_t)16384 * 256 * 2);
  W.Pq = (bf16*)take((size_t)512 * 256 * 2);
  W.Pk = (bf16*)take((size_t)512 * 256 * 2);
  W.posb = (bf16*)take((size_t)512 * 256 * 2);
  W.wbin = (bf16*)take((size_t)768 * 256 * 2);
  W.wbout = (bf16*)take((size_t)256 * 256 * 2);
  W.raw = (float*)take((size_t)64 * 8 * 256 * 256 * 4);

  // Convert weights & pos table to bf16.
  k_cvt_bf16<<<(768 * 256 + 255) / 256, 256, 0, stream>>>(in_w, W.wbin, 768 * 256);
  k_cvt_bf16<<<(256 * 256 + 255) / 256, 256, 0, stream>>>(out_w, W.wbout, 256 * 256);
  k_cvt_bf16<<<(511 * 256 + 255) / 256, 256, 0, stream>>>(pos, W.posb, 511 * 256);

  // Project the (unflipped) pos table once; flip folds into the gather index.
  k_proj_gemm<<<dim3(32, 4), 128, 0, stream>>>(W.posb, W.wbin, in_b, kScale, W.Pq, 256);
  k_proj_gemm<<<dim3(32, 4), 128, 0, stream>>>(W.posb, W.wbin + 256 * 256, in_b + 256, 1.0f,
                                               W.Pk, 256);

  // LayerNorm (ln1) of both feature maps.
  k_layernorm<<<2048, 256, 0, stream>>>(feat_left, ln1_g, ln1_b, W.lnL, 0);
  k_layernorm<<<2048, 256, 0, stream>>>(feat_right, ln1_g, ln1_b, W.lnR, 0);

  // Pass 1: right queries left (flipped pos); result -> right half of output.
  run_pass(W, W.lnR, W.lnL, /*flip=*/1, /*halfoff=*/64, feat_right, /*rawsum=*/0,
           in_b, out_b, pidx, out, stream);

  // ln2 on updated feat_right (read back from the strided output layout).
  k_layernorm<<<2048, 256, 0, stream>>>(out, ln2_g, ln2_b, W.lnT, 1);

  // Pass 2: left queries updated right; result -> left half; emit raw_attn sum.
  run_pass(W, W.lnL, W.lnT, /*flip=*/0, /*halfoff=*/0, feat_left, /*rawsum=*/1,
           in_b, out_b, pidx, out, stream);
}
